// Adaptive_Self_Attention_81140522156784
// MI455X (gfx1250) — matrix operand order repaired and checked by reading, not compiled, not run on hardware
//
#include <hip/hip_runtime.h>
#include <hip/hip_bf16.h>
#include <math.h>

// ---------------- CDNA5 WMMA types ----------------
typedef __bf16 bf16_t;
typedef __attribute__((ext_vector_type(16))) __bf16 v16bf;
typedef __attribute__((ext_vector_type(8)))  __bf16 v8bf;
typedef __attribute__((ext_vector_type(8)))  float  v8f;

static __device__ __forceinline__ unsigned bfbits(float f) {   // RNE, low 16 valid
    unsigned u = __builtin_bit_cast(unsigned, f);
    u += 0x7FFFu + ((u >> 16) & 1u);
    return u >> 16;
}
static __device__ __forceinline__ __bf16 f2b(float f) {
    unsigned short h = (unsigned short)bfbits(f);
    return __builtin_bit_cast(__bf16, h);
}
// 16 contiguous bf16 -> v16bf (two 16B loads)
static __device__ __forceinline__ v16bf ld16c(const bf16_t* p) {
    v8bf lo = *reinterpret_cast<const v8bf*>(p);
    v8bf hi = *reinterpret_cast<const v8bf*>(p + 8);
    return __builtin_shufflevector(lo, hi, 0,1,2,3,4,5,6,7,8,9,10,11,12,13,14,15);
}
// A-fragment: elements 0..15 contiguous at p (same k order as B)
static __device__ __forceinline__ v16bf ldA16(const bf16_t* p) {
    v8bf lo = *reinterpret_cast<const v8bf*>(p);
    v8bf hi = *reinterpret_cast<const v8bf*>(p + 8);
    return __builtin_shufflevector(lo, hi, 0,1,2,3,4,5,6,7,8,9,10,11,12,13,14,15);
}
static __device__ __forceinline__ float geluf(float x) {
    return 0.5f * x * (1.0f + erff(x * 0.70710678118654752f));
}
static __device__ __forceinline__ float sigm1p(float x) {   // 1 + sigmoid(x)
    return 1.0f + 1.0f / (1.0f + __expf(-x));
}

#define BB   8
#define HH   128
#define WW   128
#define CC   192
#define LL   (HH * WW)              // 16384
#define C3   (3 * CC)               // 576
#define C8   24
#define CH   96
#define HD   32
#define BN_INV 0.99999500003749969f // 1/sqrt(1+1e-5)

// =====================================================================
// Pre-conversion kernels (run once per launch; streaming, packed)
// =====================================================================
__global__ __launch_bounds__(256) void k_cvt_x(
    const float4* __restrict__ in, uint2* __restrict__ out, int n4)
{
    int i = blockIdx.x * blockDim.x + threadIdx.x;
    if (i >= n4) return;
    float4 v = in[i];
    unsigned lo = bfbits(v.x) | (bfbits(v.y) << 16);
    unsigned hi = bfbits(v.z) | (bfbits(v.w) << 16);
    out[i] = make_uint2(lo, hi);
}
// w_qkv (K=192, N=576) -> wqT (N,K) bf16
__global__ __launch_bounds__(256) void k_cvt_wq(
    const float* __restrict__ w, bf16_t* __restrict__ wt)
{
    int i = blockIdx.x * blockDim.x + threadIdx.x;   // 576*192
    if (i >= C3 * CC) return;
    int n = i / CC, k = i % CC;
    wt[i] = f2b(w[k * C3 + n]);
}
// w_proj (K=192, N=192) -> wpT (N,K) bf16
__global__ __launch_bounds__(256) void k_cvt_wp(
    const float* __restrict__ w, bf16_t* __restrict__ wt)
{
    int i = blockIdx.x * blockDim.x + threadIdx.x;   // 192*192
    if (i >= CC * CC) return;
    int n = i / CC, k = i % CC;
    wt[i] = f2b(w[k * CC + n]);
}

// =====================================================================
// Kernel 1: qkv = xb @ wqT^T + b_qkv -> bf16 (B*L, 576)
// Wave computes 4 adjacent 16x16 N-tiles; A fragment reused 4x.
// =====================================================================
__global__ __launch_bounds__(256) void k_gemm_qkv(
    const bf16_t* __restrict__ A, const bf16_t* __restrict__ WT,
    const float* __restrict__ bq, bf16_t* __restrict__ out)
{
    int wave = (int)((blockIdx.x * blockDim.x + threadIdx.x) >> 5);
    int lane = threadIdx.x & 31;
    int tg = wave % 9, tm = wave / 9;                 // 9 groups of 4 tiles = 576
    if (tm >= (BB * LL) / 16) return;
    int half16 = lane >> 4, mr = lane & 15, h8 = half16 * 8;
    int row0 = tm * 16, col0 = tg * 64;

    v8f acc[4] = {{}, {}, {}, {}};
    const bf16_t* arow = A + (size_t)(row0 + mr) * CC;
    const bf16_t* bp[4];
#pragma unroll
    for (int j = 0; j < 4; ++j) bp[j] = WT + (size_t)(col0 + j * 16 + mr) * CC;

#pragma unroll
    for (int kk = 0; kk < CC / 32; ++kk) {
        int k0 = kk * 32, kb = k0 + half16 * 16;
        v16bf a = ldA16(arow + k0 + half16 * 16);
#pragma unroll
        for (int j = 0; j < 4; ++j)
            acc[j] = __builtin_amdgcn_wmma_f32_16x16x32_bf16(
                false, a, false, ld16c(bp[j] + kb), (short)0, acc[j], false, false);
    }
#pragma unroll
    for (int j = 0; j < 4; ++j) {
        int ncol = col0 + j * 16 + mr;
        float bias = bq[ncol];
#pragma unroll
        for (int r = 0; r < 8; ++r) {
            int row = row0 + r + 8 * half16;
            out[(size_t)row * C3 + ncol] = f2b(acc[j][r] + bias);
        }
    }
}

// =====================================================================
// Kernel 2: window attention. Scores live in VGPRs; softmax via shuffles;
// P (bf16), K, V^T staged in LDS (53.8 KB total).
// =====================================================================
__global__ __launch_bounds__(256) void k_attn(
    const bf16_t* __restrict__ qkv, float* __restrict__ fu)
{
    int halfIdx = blockIdx.y;                   // 0: 8x16 windows, 1: 16x8
    int head = blockIdx.x % 3;
    int win  = blockIdx.x / 3;
    int Hs = halfIdx ? 16 : 8;
    int Ws = halfIdx ? 8 : 16;
    int nWw = WW / Ws;
    int b   = win >> 7;
    int rem = win & 127;
    int hb = rem / nWw, wb = rem % nWw;
    int cbase = halfIdx * CH + head * HD;

    int tid = threadIdx.x, lane = tid & 31, wv = tid >> 5;
    int half16 = lane >> 4, mr = lane & 15, h8 = half16 * 8;

    __shared__ bf16_t Kl[128 * 40];             // 10240 B (pad 40)
    __shared__ bf16_t Vt[32 * 136];             //  8704 B (V transposed, pad 136)
    __shared__ bf16_t Pb[128 * 136];            // 34816 B (softmax probs, pad 136)

    auto rowOf = [&](int t) -> size_t {
        int hh = hb * Hs + t / Ws;
        int ww = wb * Ws + (t % Ws);
        return (size_t)b * LL + (size_t)hh * WW + ww;
    };

    // ---- Stage K and V tiles into LDS (coalesced on d) ----
    for (int i = tid; i < 128 * 32; i += 256) {
        int t = i >> 5, d = i & 31;
        size_t base = rowOf(t) * C3 + cbase + d;
        Kl[t * 40 + d]  = qkv[base + CC];
        Vt[d * 136 + t] = qkv[base + 2 * CC];
    }
    __syncthreads();

    // ---- Phase 1: S-stripe (16 rows x 128 cols) in registers ----
    int ti = wv;
    v8f acc[8];
    {
        const bf16_t* qp = qkv + rowOf(ti * 16 + mr) * C3 + cbase;
        v16bf a = ldA16(qp + half16 * 16);
#pragma unroll
        for (int tj = 0; tj < 8; ++tj) {
            const bf16_t* kp = &Kl[(tj * 16 + mr) * 40 + half16 * 16];
            v8f z = {};
            acc[tj] = __builtin_amdgcn_wmma_f32_16x16x32_bf16(
                false, a, false, ld16c(kp), (short)0, z, false, false);
        }
    }

    // ---- Phase 2: softmax per row (16-lane shuffle reductions) ----
    {
        const float sc = 0.53033008588991064f;  // 3 / sqrt(32)
#pragma unroll
        for (int tj = 0; tj < 8; ++tj)
#pragma unroll
            for (int r = 0; r < 8; ++r) acc[tj][r] *= sc;

#pragma unroll
        for (int r = 0; r < 8; ++r) {
            float mx = acc[0][r];
#pragma unroll
            for (int tj = 1; tj < 8; ++tj) mx = fmaxf(mx, acc[tj][r]);
            mx = fmaxf(mx, __shfl_xor(mx, 1, 16));
            mx = fmaxf(mx, __shfl_xor(mx, 2, 16));
            mx = fmaxf(mx, __shfl_xor(mx, 4, 16));
            mx = fmaxf(mx, __shfl_xor(mx, 8, 16));
            float sum = 0.f;
#pragma unroll
            for (int tj = 0; tj < 8; ++tj) {
                float e = __expf(acc[tj][r] - mx);
                acc[tj][r] = e;
                sum += e;
            }
            sum += __shfl_xor(sum, 1, 16);
            sum += __shfl_xor(sum, 2, 16);
            sum += __shfl_xor(sum, 4, 16);
            sum += __shfl_xor(sum, 8, 16);
            float s3 = 3.0f / sum;               // fold the 3*v factor
            int row = ti * 16 + r + 8 * half16;
#pragma unroll
            for (int tj = 0; tj < 8; ++tj)
                Pb[row * 136 + tj * 16 + mr] = f2b(acc[tj][r] * s3);
        }
    }
    __syncthreads();

    // ---- Phase 3: O = P @ V ----
#pragma unroll
    for (int tn = 0; tn < 2; ++tn) {
        v8f o = {};
#pragma unroll
        for (int ks = 0; ks < 4; ++ks) {
            int k0 = ks * 32;
            v16bf a = ldA16(&Pb[(ti * 16 + mr) * 136 + k0 + half16 * 16]);
            const bf16_t* vp = &Vt[(tn * 16 + mr) * 136 + k0 + half16 * 16];
            o = __builtin_amdgcn_wmma_f32_16x16x32_bf16(
                false, a, false, ld16c(vp), (short)0, o, false, false);
        }
#pragma unroll
        for (int r = 0; r < 8; ++r) {
            size_t orow = rowOf(ti * 16 + r + 8 * half16);
            fu[orow * CC + cbase + tn * 16 + mr] = o[r];
        }
    }
}

// =====================================================================
// Kernel 3: depthwise 3x3 + BN + GELU on raw-reshaped x -> conv1 (B,C,H,W)
// =====================================================================
__global__ __launch_bounds__(256) void k_dwconv1(
    const float* __restrict__ in, const float* __restrict__ wk,
    const float* __restrict__ bias, const float* __restrict__ g,
    const float* __restrict__ be, float* __restrict__ out)
{
    size_t idx = (size_t)blockIdx.x * blockDim.x + threadIdx.x;
    if (idx >= (size_t)BB * CC * LL) return;
    int wx = idx & (WW - 1);
    int hy = (idx >> 7) & (HH - 1);
    int c  = (int)((idx >> 14) % CC);
    int bb = (int)(idx / ((size_t)CC * LL));
    const float* base = in + (size_t)bb * CC * LL + (size_t)c * LL;
    const float* kw = wk + c * 9;
    float s = bias[c];
#pragma unroll
    for (int dy = -1; dy <= 1; ++dy) {
        int h2 = hy + dy;
        if (h2 < 0 || h2 >= HH) continue;
#pragma unroll
        for (int dx = -1; dx <= 1; ++dx) {
            int w2 = wx + dx;
            if (w2 < 0 || w2 >= WW) continue;
            s += base[h2 * WW + w2] * kw[(dy + 1) * 3 + (dx + 1)];
        }
    }
    out[idx] = geluf(s * (g[c] * BN_INV) + be[c]);
}

// =====================================================================
// Kernel 4: spatial map
// =====================================================================
__global__ __launch_bounds__(256) void k_spatial(
    const float* __restrict__ conv1, const float* __restrict__ w1,
    const float* __restrict__ b1, const float* __restrict__ gsi,
    const float* __restrict__ besi, const float* __restrict__ w2,
    const float* __restrict__ b2, float* __restrict__ sp1)
{
    int p = blockIdx.x * blockDim.x + threadIdx.x;      // B*L
    if (p >= BB * LL) return;
    int b = p >> 14, hw = p & (LL - 1);
    float acc[C8];
#pragma unroll
    for (int o = 0; o < C8; ++o) acc[o] = b1[o];
    const float* base = conv1 + (size_t)b * CC * LL + hw;
    for (int c = 0; c < CC; ++c) {
        float v = base[(size_t)c * LL];
#pragma unroll
        for (int o = 0; o < C8; ++o) acc[o] += v * w1[o * CC + c];
    }
    float s = b2[0];
#pragma unroll
    for (int o = 0; o < C8; ++o)
        s += geluf(acc[o] * (gsi[o] * BN_INV) + besi[o]) * w2[o];
    sp1[p] = sigm1p(s);
}

// =====================================================================
// Kernel 5a/5b: channel map
// =====================================================================
__global__ __launch_bounds__(256) void k_mean(
    const float* __restrict__ conv1, float* __restrict__ means)
{
    int bc = blockIdx.x;
    const float* base = conv1 + (size_t)bc * LL;
    float s = 0.f;
    for (int i = threadIdx.x; i < LL; i += 256) s += base[i];
    __shared__ float red[256];
    red[threadIdx.x] = s;
    __syncthreads();
    for (int st = 128; st > 0; st >>= 1) {
        if (threadIdx.x < st) red[threadIdx.x] += red[threadIdx.x + st];
        __syncthreads();
    }
    if (threadIdx.x == 0) means[bc] = red[0] * (1.0f / (float)LL);
}

__global__ __launch_bounds__(256) void k_channel(
    const float* __restrict__ means, const float* __restrict__ w1,
    const float* __restrict__ b1, const float* __restrict__ gci,
    const float* __restrict__ beci, const float* __restrict__ w2,
    const float* __restrict__ b2, float* __restrict__ chs)
{
    int b = blockIdx.x, t = threadIdx.x;
    __shared__ float t1[C8];
    if (t < C8) {
        float s = b1[t];
        for (int c = 0; c < CC; ++c) s += means[b * CC + c] * w1[t * CC + c];
        t1[t] = geluf(s * (gci[t] * BN_INV) + beci[t]);
    }
    __syncthreads();
    if (t < CC) {
        float s = b2[t];
#pragma unroll
        for (int o = 0; o < C8; ++o) s += t1[o] * w2[t * C8 + o];
        chs[b * CC + t] = sigm1p(s);
    }
}

// =====================================================================
// Kernel 6a: spatial gate + transpose to NCHW
// =====================================================================
__global__ __launch_bounds__(256) void k_fuse_spatial(
    const float* __restrict__ fu, const float* __restrict__ sp1,
    float* __restrict__ fu_s)
{
    size_t idx = (size_t)blockIdx.x * blockDim.x + threadIdx.x;
    if (idx >= (size_t)BB * CC * LL) return;
    int hw = (int)(idx & (LL - 1));
    int c  = (int)((idx >> 14) % CC);
    int b  = (int)(idx / ((size_t)CC * LL));
    fu_s[idx] = fu[((size_t)b * LL + hw) * CC + c] * sp1[b * LL + hw];
}

// =====================================================================
// Kernel 6b: dwconv2 + BN + GELU + channel gate -> bf16 (B,L,C)
// =====================================================================
__global__ __launch_bounds__(256) void k_dwconv2(
    const float* __restrict__ in, const float* __restrict__ wk,
    const float* __restrict__ bias, const float* __restrict__ g,
    const float* __restrict__ be, const float* __restrict__ chs,
    bf16_t* __restrict__ outb)
{
    size_t idx = (size_t)blockIdx.x * blockDim.x + threadIdx.x;
    if (idx >= (size_t)BB * CC * LL) return;
    int wx = idx & (WW - 1);
    int hy = (idx >> 7) & (HH - 1);
    int c  = (int)((idx >> 14) % CC);
    int bb = (int)(idx / ((size_t)CC * LL));
    const float* base = in + (size_t)bb * CC * LL + (size_t)c * LL;
    const float* kw = wk + c * 9;
    float s = bias[c];
#pragma unroll
    for (int dy = -1; dy <= 1; ++dy) {
        int h2 = hy + dy;
        if (h2 < 0 || h2 >= HH) continue;
#pragma unroll
        for (int dx = -1; dx <= 1; ++dx) {
            int w2 = wx + dx;
            if (w2 < 0 || w2 >= WW) continue;
            s += base[h2 * WW + w2] * kw[(dy + 1) * 3 + (dx + 1)];
        }
    }
    float v = geluf(s * (g[c] * BN_INV) + be[c]) * chs[bb * CC + c];
    int hw = hy * WW + wx;
    outb[((size_t)bb * LL + hw) * CC + c] = f2b(v);
}

// =====================================================================
// Kernel 7: out = fuB @ wpT^T + b_proj (f32 out). 4 N-tiles per wave.
// =====================================================================
__global__ __launch_bounds__(256) void k_gemm_proj(
    const bf16_t* __restrict__ A, const bf16_t* __restrict__ WT,
    const float* __restrict__ bp, float* __restrict__ D)
{
    int wave = (int)((blockIdx.x * blockDim.x + threadIdx.x) >> 5);
    int lane = threadIdx.x & 31;
    int tg = wave % 3, tm = wave / 3;                 // 3 groups of 4 tiles = 192
    if (tm >= (BB * LL) / 16) return;
    int half16 = lane >> 4, mr = lane & 15, h8 = half16 * 8;
    int row0 = tm * 16, col0 = tg * 64;

    v8f acc[4] = {{}, {}, {}, {}};
    const bf16_t* arow = A + (size_t)(row0 + mr) * CC;
    const bf16_t* bpp[4];
#pragma unroll
    for (int j = 0; j < 4; ++j) bpp[j] = WT + (size_t)(col0 + j * 16 + mr) * CC;

#pragma unroll
    for (int kk = 0; kk < CC / 32; ++kk) {
        int k0 = kk * 32, kb = k0 + half16 * 16;
        v16bf a = ldA16(arow + k0 + half16 * 16);
#pragma unroll
        for (int j = 0; j < 4; ++j)
            acc[j] = __builtin_amdgcn_wmma_f32_16x16x32_bf16(
                false, a, false, ld16c(bpp[j] + kb), (short)0, acc[j], false, false);
    }
#pragma unroll
    for (int j = 0; j < 4; ++j) {
        int ncol = col0 + j * 16 + mr;
        float bias = bp[ncol];
#pragma unroll
        for (int r = 0; r < 8; ++r) {
            int row = row0 + r + 8 * half16;
            D[(size_t)row * CC + ncol] = acc[j][r] + bias;
        }
    }
}

// =====================================================================
extern "C" void kernel_launch(void* const* d_in, const int* in_sizes, int n_in,
                              void* d_out, int out_size, void* d_ws, size_t ws_size,
                              hipStream_t stream)
{
    const float* x      = (const float*)d_in[0];
    const float* w_qkv  = (const float*)d_in[1];
    const float* b_qkv  = (const float*)d_in[2];
    const float* w_dw1  = (const float*)d_in[3];
    const float* b_dw1  = (const float*)d_in[4];
    const float* g_bn1  = (const float*)d_in[5];
    const float* be_bn1 = (const float*)d_in[6];
    const float* w_si1  = (const float*)d_in[7];
    const float* b_si1  = (const float*)d_in[8];
    const float* g_si   = (const float*)d_in[9];
    const float* be_si  = (const float*)d_in[10];
    const float* w_si2  = (const float*)d_in[11];
    const float* b_si2  = (const float*)d_in[12];
    const float* w_ci1  = (const float*)d_in[13];
    const float* b_ci1  = (const float*)d_in[14];
    const float* g_ci   = (const float*)d_in[15];
    const float* be_ci  = (const float*)d_in[16];
    const float* w_ci2  = (const float*)d_in[17];
    const float* b_ci2  = (const float*)d_in[18];
    const float* w_dw2  = (const float*)d_in[19];
    const float* b_dw2  = (const float*)d_in[20];
    const float* g_bn2  = (const float*)d_in[21];
    const float* be_bn2 = (const float*)d_in[22];
    const float* w_proj = (const float*)d_in[23];
    const float* b_proj = (const float*)d_in[24];

    char* ws = (char*)d_ws;
    const size_t SZ_QKV  = (size_t)BB * LL * C3 * sizeof(bf16_t);   // 151 MB
    const size_t SZ_FU   = (size_t)BB * LL * CC * sizeof(float);    // 100 MB
    const size_t SZ_CONV = (size_t)BB * CC * LL * sizeof(float);    // 100 MB
    const size_t SZ_XB   = (size_t)BB * LL * CC * sizeof(bf16_t);   //  50 MB

    bf16_t* qkv   = (bf16_t*)ws;
    float*  fu    = (float*)(ws + SZ_QKV);
    float*  conv1 = (float*)(ws + SZ_QKV + SZ_FU);
    bf16_t* xb    = (bf16_t*)(ws + SZ_QKV + SZ_FU + SZ_CONV);
    char*   tail  = ws + SZ_QKV + SZ_FU + SZ_CONV + SZ_XB;
    bf16_t* wqT   = (bf16_t*)tail;                       // 576*192*2
    bf16_t* wpT   = (bf16_t*)(tail + (size_t)C3 * CC * 2);
    float*  sp1   = (float*)(tail + (size_t)C3 * CC * 2 + (size_t)CC * CC * 2);
    float*  means = sp1 + (size_t)BB * LL;
    float*  chs   = means + BB * CC;
    float*  fu_s  = (float*)ws;                          // reuse qkv region
    bf16_t* fuB   = (bf16_t*)conv1;                      // reuse conv1 region

    const int TPB = 256;
    const int n4 = (BB * LL * CC) / 4;                   // 6291456
    // 0. pre-convert activations and weights to bf16
    k_cvt_x<<<n4 / TPB, TPB, 0, stream>>>((const float4*)x, (uint2*)xb, n4);
    k_cvt_wq<<<(C3 * CC + TPB - 1) / TPB, TPB, 0, stream>>>(w_qkv, wqT);
    k_cvt_wp<<<(CC * CC + TPB - 1) / TPB, TPB, 0, stream>>>(w_proj, wpT);
    // 1. QKV GEMM: 8192 row-tiles * 9 groups = 73728 waves -> 9216 blocks
    k_gemm_qkv<<<9216, TPB, 0, stream>>>(xb, wqT, b_qkv, qkv);
    // 2. depthwise conv 1
    k_dwconv1<<<(int)(((size_t)BB * CC * LL) / TPB), TPB, 0, stream>>>(
        x, w_dw1, b_dw1, g_bn1, be_bn1, conv1);
    // 3. window attention
    k_attn<<<dim3(3072, 2), TPB, 0, stream>>>(qkv, fu);
    // 4. spatial interaction
    k_spatial<<<(BB * LL) / TPB, TPB, 0, stream>>>(
        conv1, w_si1, b_si1, g_si, be_si, w_si2, b_si2, sp1);
    // 5. channel interaction
    k_mean<<<BB * CC, TPB, 0, stream>>>(conv1, means);
    k_channel<<<BB, TPB, 0, stream>>>(means, w_ci1, b_ci1, g_ci, be_ci, w_ci2, b_ci2, chs);
    // 6. gates + dwconv2
    k_fuse_spatial<<<(int)(((size_t)BB * CC * LL) / TPB), TPB, 0, stream>>>(fu, sp1, fu_s);
    k_dwconv2<<<(int)(((size_t)BB * CC * LL) / TPB), TPB, 0, stream>>>(
        fu_s, w_dw2, b_dw2, g_bn2, be_bn2, chs, fuB);
    // 7. final projection: 8192 * 3 = 24576 waves -> 3072 blocks
    k_gemm_proj<<<3072, TPB, 0, stream>>>(fuB, wpT, b_proj, (float*)d_out);
}
